// GCN_65000035058580
// MI455X (gfx1250) — matrix operand order repaired and checked by reading, not compiled, not run on hardware
//
#include <hip/hip_runtime.h>

#define N_NODES 50000
#define N_EDGES 800000
#define IN_CH   100
#define H_CH    128
#define N_GRAPH 512
#define LN_EPS  1e-5f

typedef __attribute__((ext_vector_type(8)))  float  v8f;
typedef __attribute__((ext_vector_type(8)))  __bf16 v8bf;
typedef __attribute__((ext_vector_type(16))) __bf16 v16bf;
typedef __attribute__((ext_vector_type(4)))  int    v4i;

#if __has_builtin(__builtin_amdgcn_global_load_async_to_lds_b128) && \
    __has_builtin(__builtin_amdgcn_s_wait_asynccnt)
#define USE_ASYNC_LDS 1
typedef __attribute__((address_space(1))) v4i gv4i;   // global v4i
typedef __attribute__((address_space(3))) v4i lv4i;   // LDS v4i
#endif

__device__ __forceinline__ __bf16 f2bf(float f) {
  unsigned u = __builtin_bit_cast(unsigned, f);
  unsigned r = (u + 0x7FFFu + ((u >> 16) & 1u)) >> 16;   // round-to-nearest-even
  unsigned short s = (unsigned short)r;
  return __builtin_bit_cast(__bf16, s);
}

// ---------------------------------------------------------------- fill
__global__ __launch_bounds__(256) void fill_zero(float* __restrict__ p, int n) {
  int i = blockIdx.x * 256 + threadIdx.x;
  if (i < n) p[i] = 0.f;
}

// ---------------------------------------------------------------- weight packing into WMMA B-fragment layout
// Wpack layout: [ntile(8)][kstep(4)][lane(32)][elem(16)] bf16, contiguous per lane.
// B (32x16, 16-bit): lane half h holds K=8h..8h+7 (elems 0-7) and K=16+8h..16+8h+7 (elems 8-15), col N=lane&15.
__global__ __launch_bounds__(256) void pack_w(const float* __restrict__ W, int K,
                                              __bf16* __restrict__ out) {
  int t = blockIdx.x * 256 + threadIdx.x;   // 0..1023 = 8*4*32
  if (t >= 8 * 4 * 32) return;
  int lane = t & 31;
  int half = lane >> 4, idx = lane & 15;
  int nt = t >> 7;
  int n  = nt * 16 + idx;
  __bf16* o = out + (size_t)t * 16;
#pragma unroll
  for (int e = 0; e < 16; ++e) {
    int k = ((t >> 5) & 3) * 32 + half * 8 + (e < 8 ? e : e + 8);
    float v = (k < K) ? W[(size_t)k * H_CH + n] : 0.f;
    o[e] = f2bf(v);
  }
}

// ---------------------------------------------------------------- LayerNorm -> bf16 (zero padded to 128)
__global__ __launch_bounds__(256) void ln_to_bf16(const float* __restrict__ h, int ld, int K,
                                                  const float* __restrict__ gam,
                                                  const float* __restrict__ bet,
                                                  __bf16* __restrict__ out) {
  int node = blockIdx.x * 8 + (threadIdx.x >> 5);
  int lane = threadIdx.x & 31;
  const float* row = h + (size_t)node * ld;
  float v[4];
  float s = 0.f;
#pragma unroll
  for (int j = 0; j < 4; ++j) {
    int k = j * 32 + lane;
    v[j] = (k < K) ? row[k] : 0.f;
    s += v[j];
  }
#pragma unroll
  for (int off = 16; off > 0; off >>= 1) s += __shfl_xor(s, off, 32);
  float mean = s / (float)K;
  float q = 0.f;
#pragma unroll
  for (int j = 0; j < 4; ++j) {
    int k = j * 32 + lane;
    float d = (k < K) ? (v[j] - mean) : 0.f;
    q += d * d;
  }
#pragma unroll
  for (int off = 16; off > 0; off >>= 1) q += __shfl_xor(q, off, 32);
  float rs = rsqrtf(q / (float)K + LN_EPS);
  __bf16* orow = out + (size_t)node * H_CH;
#pragma unroll
  for (int j = 0; j < 4; ++j) {
    int k = j * 32 + lane;
    float val = (k < K) ? ((v[j] - mean) * rs * gam[k] + bet[k]) : 0.f;
    orow[k] = f2bf(val);
  }
}

// ---------------------------------------------------------------- WMMA GEMM: [N_NODES x 128] bf16 @ [128 x 128] bf16 -> fp32
// Block = 256 threads = 8 waves. Packed weights (32 KB) staged to LDS once per
// block (async global->LDS when available), then wave w computes m-tile
// (blockIdx*4 + w/2), n-half (w&1): 4 accumulators, 16 WMMAs.
__global__ __launch_bounds__(256) void gemm_bf16_wmma(const __bf16* __restrict__ A,
                                                      const __bf16* __restrict__ Wp,
                                                      float* __restrict__ out) {
  __shared__ __bf16 sW[8 * 4 * 32 * 16];   // 16384 bf16 = 32 KB

  int tid = threadIdx.x;
  // Stage 32768 B with 256 threads x 16 B x 8 iterations.
#if defined(USE_ASYNC_LDS)
#pragma unroll
  for (int i = 0; i < 8; ++i) {
    int off = (i * 256 + tid) * 8;   // bf16 elements, 16 B chunks
    __builtin_amdgcn_global_load_async_to_lds_b128(
        (gv4i*)(Wp + off), (lv4i*)(sW + off), 0, 0);
  }
  __builtin_amdgcn_s_wait_asynccnt(0);
  __syncthreads();
#else
#pragma unroll
  for (int i = 0; i < 8; ++i) {
    int off = (i * 256 + tid) * 8;
    *(v8bf*)(sW + off) = *(const v8bf*)(Wp + off);
  }
  __syncthreads();
#endif

  int wave = tid >> 5;
  int lane = tid & 31;
  int half = lane >> 4, idx = lane & 15;
  int mt = blockIdx.x * 4 + (wave >> 1);
  int ng = wave & 1;                       // n-half: tiles 4*ng .. 4*ng+3
  if (mt >= N_NODES / 16) return;          // after barrier: safe

  // A fragment base (ISA 16-bit A 16x32 layout): lane half selects K sub-range.
  const __bf16* arow = A + (size_t)(mt * 16 + idx) * H_CH + half * 8;

  v8f acc[4];
#pragma unroll
  for (int u = 0; u < 4; ++u) acc[u] = (v8f){};

#pragma unroll
  for (int ks = 0; ks < 4; ++ks) {
    v8bf lo = *(const v8bf*)(arow + ks * 32);        // K = ks*32 + half*8 + [0..7]
    v8bf hi = *(const v8bf*)(arow + ks * 32 + 16);   // K = ks*32 + 16 + half*8 + [0..7]
    v16bf a;
#pragma unroll
    for (int i = 0; i < 8; ++i) { a[i] = lo[i]; a[i + 8] = hi[i]; }
#pragma unroll
    for (int u = 0; u < 4; ++u) {
      int nt = ng * 4 + u;
      v16bf b = *(const v16bf*)(sW + (((nt * 4 + ks)) * 32 + lane) * 16);
      acc[u] = __builtin_amdgcn_wmma_f32_16x16x32_bf16(
          /*neg_a=*/false, a, /*neg_b=*/false, b,
          /*c_mod=*/(short)0, acc[u], /*reuse_a=*/false, /*reuse_b=*/false);
    }
  }

  // C/D layout: VGPR r -> row (half*8 + r), col = idx within the 16x16 tile.
#pragma unroll
  for (int u = 0; u < 4; ++u) {
    int n = (ng * 4 + u) * 16 + idx;
#pragma unroll
    for (int r = 0; r < 8; ++r) {
      out[(size_t)(mt * 16 + half * 8 + r) * H_CH + n] = acc[u][r];
    }
  }
}

// ---------------------------------------------------------------- edge scatter: agg[dst] += hW[src] * w
// One wave32 per edge: indices/weight are wave-uniform (scalarized loads),
// each lane handles 4 features via one float4 gather + 4 f32 atomics.
__global__ __launch_bounds__(256) void scatter_edges(const float* __restrict__ hw,
                                                     const long long* __restrict__ ei,
                                                     const float* __restrict__ ew,
                                                     float* __restrict__ agg) {
  int e    = blockIdx.x * 8 + (threadIdx.x >> 5);
  int lane = threadIdx.x & 31;
  int src = (int)ei[e];
  int dst = (int)ei[N_EDGES + e];
  float w = ew[e];
  const float4 v = *(const float4*)(hw + (size_t)src * H_CH + lane * 4);
  float* base = agg + (size_t)dst * H_CH + lane * 4;
  atomicAdd(base + 0, v.x * w);
  atomicAdd(base + 1, v.y * w);
  atomicAdd(base + 2, v.z * w);
  atomicAdd(base + 3, v.w * w);
}

// ---------------------------------------------------------------- bias + relu
__global__ __launch_bounds__(256) void bias_relu(const float* __restrict__ agg,
                                                 const float* __restrict__ bias,
                                                 float* __restrict__ out) {
  int t = blockIdx.x * 256 + threadIdx.x;
  int f = t & 127;
  float v = agg[t] + bias[f];
  out[t] = v > 0.f ? v : 0.f;
}

// ---------------------------------------------------------------- pooling
__global__ __launch_bounds__(256) void pool_accum(const float* __restrict__ h,
                                                  const long long* __restrict__ batch,
                                                  float* __restrict__ psum) {
  int t = blockIdx.x * 256 + threadIdx.x;
  int n = t >> 7, f = t & 127;
  int g = (int)batch[n];
  atomicAdd(psum + (size_t)g * H_CH + f, h[t]);
}

__global__ __launch_bounds__(256) void pool_count(const long long* __restrict__ batch,
                                                  float* __restrict__ pcnt) {
  int n = blockIdx.x * 256 + threadIdx.x;
  if (n < N_NODES) atomicAdd(pcnt + (int)batch[n], 1.0f);
}

// ---------------------------------------------------------------- classifier: pooled @ Wlin + blin
__global__ __launch_bounds__(256) void classify(const float* __restrict__ psum,
                                                const float* __restrict__ pcnt,
                                                const float* __restrict__ Wlin,
                                                const float* __restrict__ blin,
                                                float* __restrict__ out) {
  int g = blockIdx.x * 8 + (threadIdx.x >> 5);
  int lane = threadIdx.x & 31;
  float cnt = pcnt[g];
  cnt = cnt > 1.f ? cnt : 1.f;
  float inv = 1.f / cnt;
  float p0 = 0.f, p1 = 0.f;
#pragma unroll
  for (int j = 0; j < 4; ++j) {
    int k = j * 32 + lane;
    float pv = psum[(size_t)g * H_CH + k] * inv;
    p0 += pv * Wlin[k * 2 + 0];
    p1 += pv * Wlin[k * 2 + 1];
  }
#pragma unroll
  for (int off = 16; off > 0; off >>= 1) {
    p0 += __shfl_xor(p0, off, 32);
    p1 += __shfl_xor(p1, off, 32);
  }
  if (lane == 0) {
    out[g * 2 + 0] = p0 + blin[0];
    out[g * 2 + 1] = p1 + blin[1];
  }
}

// ---------------------------------------------------------------- launch
extern "C" void kernel_launch(void* const* d_in, const int* in_sizes, int n_in,
                              void* d_out, int out_size, void* d_ws, size_t ws_size,
                              hipStream_t stream) {
  (void)in_sizes; (void)n_in; (void)out_size; (void)ws_size;
  const float*     x     = (const float*)d_in[0];
  const long long* ei    = (const long long*)d_in[1];
  const float*     ew    = (const float*)d_in[2];
  const long long* batch = (const long long*)d_in[3];
  const float* lng[3]  = {(const float*)d_in[4],  (const float*)d_in[6],  (const float*)d_in[8]};
  const float* lnb[3]  = {(const float*)d_in[5],  (const float*)d_in[7],  (const float*)d_in[9]};
  const float* W[3]    = {(const float*)d_in[10], (const float*)d_in[12], (const float*)d_in[14]};
  const float* bias[3] = {(const float*)d_in[11], (const float*)d_in[13], (const float*)d_in[15]};
  const float* Wlin = (const float*)d_in[16];
  const float* blin = (const float*)d_in[17];
  float* out = (float*)d_out;

  char* ws = (char*)d_ws;
  size_t off = 0;
  auto carve = [&](size_t bytes) -> void* {
    void* p = ws + off;
    off += (bytes + 255) & ~(size_t)255;
    return p;
  };
  float*  bufH   = (float*)carve((size_t)N_NODES * H_CH * 4);
  float*  bufHW  = (float*)carve((size_t)N_NODES * H_CH * 4);
  float*  bufAgg = (float*)carve((size_t)N_NODES * H_CH * 4);
  __bf16* nrm    = (__bf16*)carve((size_t)N_NODES * H_CH * 2);
  __bf16* wp[3];
  for (int l = 0; l < 3; ++l) wp[l] = (__bf16*)carve((size_t)8 * 4 * 32 * 16 * 2);
  float* psum = (float*)carve((size_t)N_GRAPH * H_CH * 4);
  float* pcnt = (float*)carve((size_t)N_GRAPH * 4);

  for (int l = 0; l < 3; ++l)
    pack_w<<<4, 256, 0, stream>>>(W[l], l == 0 ? IN_CH : H_CH, wp[l]);

  const int mtiles = N_NODES / 16;                 // 3125
  const int gemm_blocks = (mtiles + 3) / 4;        // 782

  const float* hin = x;
  int ld = IN_CH, K = IN_CH;
  for (int l = 0; l < 3; ++l) {
    ln_to_bf16<<<N_NODES / 8, 256, 0, stream>>>(hin, ld, K, lng[l], lnb[l], nrm);
    gemm_bf16_wmma<<<gemm_blocks, 256, 0, stream>>>(nrm, wp[l], bufHW);
    fill_zero<<<(N_NODES * H_CH) / 256, 256, 0, stream>>>(bufAgg, N_NODES * H_CH);
    scatter_edges<<<N_EDGES / 8, 256, 0, stream>>>(bufHW, ei, ew, bufAgg);
    bias_relu<<<(N_NODES * H_CH) / 256, 256, 0, stream>>>(bufAgg, bias[l], bufH);
    hin = bufH; ld = H_CH; K = H_CH;
  }

  fill_zero<<<(N_GRAPH * H_CH) / 256, 256, 0, stream>>>(psum, N_GRAPH * H_CH);
  fill_zero<<<(N_GRAPH + 255) / 256, 256, 0, stream>>>(pcnt, N_GRAPH);
  pool_accum<<<(N_NODES * H_CH) / 256, 256, 0, stream>>>(bufH, batch, psum);
  pool_count<<<(N_NODES + 255) / 256, 256, 0, stream>>>(batch, pcnt);
  classify<<<N_GRAPH / 8, 256, 0, stream>>>(psum, pcnt, Wlin, blin, out);
}
